// MessagePackBlock_23424751632780
// MI455X (gfx1250) — matrix operand order repaired and checked by reading, not compiled, not run on hardware
//
#include <hip/hip_runtime.h>

// ---------------------------------------------------------------------------
// Fused equivariant message block for MI455X (gfx1250, wave32, WMMA bf16).
// M-dim = edges (16 per wave). All matmuls via v_wmma_f32_16x16x32_bf16.
// Weights prepacked (f32 -> bf16, B-operand lane layout) into d_ws once per
// launch; main kernel streams per-edge inputs and stages only C->A transposes
// through per-wave LDS. Round 3: hoist A-panel LDS loads out of N-tile loops,
// no-scale A builds, B-pair preloads (was valu=2870 ds=730).
// ---------------------------------------------------------------------------

typedef __attribute__((ext_vector_type(16))) __bf16 v16bf;
typedef __attribute__((ext_vector_type(8)))  float  v8f;

#define DEVFN __device__ __forceinline__

#define BETA_C      1.6791767f
#define INV8        0.125f
#define INV_SQRT128 0.08838834764831845f
#define INV_SQRT32  0.17677669529663687f
#define INV_SQRT3   0.5773502691896258f

// Tile-base offsets into prepacked weights (units of 512-element tiles).
#define TB_Wn_ss 0
#define TB_Wn_vv 16
#define TB_Wn_sv 24
#define TB_Wn_vs 32
#define TB_We_ss 36
#define TB_We_vv 44
#define TB_We_sv 48
#define TB_We_vs 52
#define TB_Wns_s 54
#define TB_Wns_v 70
#define TB_Wes_s 74
#define TB_Wes_v 90
#define TB_Wno_s 94
#define TB_Wno_v 102
#define TB_Weo_s 104
#define TB_Weo_v 112
#define TB_Wnm1  114
#define TB_Wnm2  122
#define TB_Wnm3  130
#define TB_Wem1  154
#define TB_Wem2  162
#define TB_Wem3  170

struct Q2 { uint4 a, b; };

DEVFN v8f v8zero() { v8f c = {0.f,0.f,0.f,0.f,0.f,0.f,0.f,0.f}; return c; }

DEVFN unsigned short f2bfbits(float f) {
  __bf16 b = (__bf16)f;
  return __builtin_bit_cast(unsigned short, b);
}

// Pack two f32 -> packed bf16 pair (one VOP3 instruction).
DEVFN unsigned pk_cvt(float a, float b) {
  unsigned r;
  asm("v_cvt_pk_bf16_f32 %0, %1, %2" : "=v"(r) : "v"(a), "v"(b));
  return r;
}
// Packed bf16 multiply (VOP3P).
DEVFN unsigned pk_mul(unsigned a, unsigned b) {
  unsigned r;
  asm("v_pk_mul_bf16 %0, %1, %2" : "=v"(r) : "v"(a), "v"(b));
  return r;
}

DEVFN v8f wmmab(v16bf a, v16bf b, v8f c) {
  return __builtin_amdgcn_wmma_f32_16x16x32_bf16(false, a, false, b, (short)0, c, false, false);
}

// Wave-local LDS RAW fence (DS ops are in-order per wave; only data return
// needs waiting).
DEVFN void wfence() {
  __builtin_amdgcn_wave_barrier();
  asm volatile("s_wait_dscnt 0" ::: "memory");
  __builtin_amdgcn_wave_barrier();
}

// B tile (32x16) from prepacked global weights: 32 contiguous bytes per lane.
DEVFN v16bf loadB(const uint4* wq, int tilebase, int ntiles, int kt, int nt, int lane) {
  const uint4* p = wq + (size_t)(tilebase + kt * ntiles + nt) * 64 + lane * 2;
  Q2 q; q.a = p[0]; q.b = p[1];
  return __builtin_bit_cast(v16bf, q);
}

// A chunk (16 rows x 32 K) from row-major bf16 LDS staging.
DEVFN v16bf ldsA(const __bf16* buf, int stride, int r, int k0, int hi) {
  const uint4* p = (const uint4*)(buf + r * stride + k0 + hi * 8);
  Q2 q; q.a = p[0]; q.b = p[2];   // elements 0..7 then +16
  return __builtin_bit_cast(v16bf, q);
}

// Build one A chunk from 8+8 consecutive f32 (at p[0..7] and p[16..23]).
DEVFN v16bf buildA8(const float* p, float s) {
  const float4* q = (const float4*)p;
  float4 a0 = q[0], a1 = q[1], b0 = q[4], b1 = q[5];
  v16bf A;
  A[0]=(__bf16)(a0.x*s); A[1]=(__bf16)(a0.y*s); A[2]=(__bf16)(a0.z*s); A[3]=(__bf16)(a0.w*s);
  A[4]=(__bf16)(a1.x*s); A[5]=(__bf16)(a1.y*s); A[6]=(__bf16)(a1.z*s); A[7]=(__bf16)(a1.w*s);
  A[8]=(__bf16)(b0.x*s); A[9]=(__bf16)(b0.y*s); A[10]=(__bf16)(b0.z*s); A[11]=(__bf16)(b0.w*s);
  A[12]=(__bf16)(b1.x*s); A[13]=(__bf16)(b1.y*s); A[14]=(__bf16)(b1.z*s); A[15]=(__bf16)(b1.w*s);
  return A;
}
// No-scale variant (avoids identity multiplies).
DEVFN v16bf buildA8n(const float* p) {
  const float4* q = (const float4*)p;
  float4 a0 = q[0], a1 = q[1], b0 = q[4], b1 = q[5];
  v16bf A;
  A[0]=(__bf16)a0.x; A[1]=(__bf16)a0.y; A[2]=(__bf16)a0.z; A[3]=(__bf16)a0.w;
  A[4]=(__bf16)a1.x; A[5]=(__bf16)a1.y; A[6]=(__bf16)a1.z; A[7]=(__bf16)a1.w;
  A[8]=(__bf16)b0.x; A[9]=(__bf16)b0.y; A[10]=(__bf16)b0.z; A[11]=(__bf16)b0.w;
  A[12]=(__bf16)b1.x; A[13]=(__bf16)b1.y; A[14]=(__bf16)b1.z; A[15]=(__bf16)b1.w;
  return A;
}

// 24 consecutive f32 (3-comp vectors for 8 u's) -> dot chunk + 3 per-comp chunks.
DEVFN void buildV(const float* pv, int ustart, float y0, float y1, float y2,
                  float dsc, float vsc, int h,
                  v16bf& ad, v16bf& a0, v16bf& a1, v16bf& a2) {
  const float4* q = (const float4*)(pv + 3 * ustart);
  float x[24];
#pragma unroll
  for (int t = 0; t < 6; ++t) { float4 f = q[t]; x[4*t]=f.x; x[4*t+1]=f.y; x[4*t+2]=f.z; x[4*t+3]=f.w; }
#pragma unroll
  for (int j = 0; j < 8; ++j) {
    float v0 = x[3*j], v1 = x[3*j+1], v2 = x[3*j+2];
    ad[h*8+j] = (__bf16)((v0*y0 + v1*y1 + v2*y2) * dsc);
    a0[h*8+j] = (__bf16)(v0 * vsc);
    a1[h*8+j] = (__bf16)(v1 * vsc);
    a2[h*8+j] = (__bf16)(v2 * vsc);
  }
}

// Store half-packed bf16 pair to two consecutive LDS rows.
DEVFN void storePair(__bf16* buf, int stride, int col, int row0, unsigned m) {
  *(unsigned short*)(buf + row0 * stride + col)       = (unsigned short)m;
  *(unsigned short*)(buf + (row0 + 1) * stride + col) = (unsigned short)(m >> 16);
}

// Store a C tile (f32 accum) to row-major bf16 LDS (pair-packed conversion).
DEVFN void storeC(__bf16* buf, int stride, int col, int hi8, v8f c, float s) {
#pragma unroll
  for (int k = 0; k < 4; ++k) {
    unsigned m = pk_cvt(c[2*k] * s, c[2*k+1] * s);
    storePair(buf, stride, col, hi8 + 2*k, m);
  }
}

// Store a C tile gated elementwise by packed-bf16 w tile (packed bf16 mul).
DEVFN void gateStore(__bf16* buf, int stride, int col, int hi8, v8f c, const unsigned* wqk) {
#pragma unroll
  for (int k = 0; k < 4; ++k) {
    unsigned m = pk_mul(pk_cvt(c[2*k], c[2*k+1]), wqk[k]);
    storePair(buf, stride, col, hi8 + 2*k, m);
  }
}

DEVFN float silu_beta(float x) {
  return BETA_C * x * __builtin_amdgcn_rcpf(1.0f + __expf(-x));
}

// 2-layer silu MLP + projection; output packed bf16 in wpk[12][4] C-layout.
DEVFN void run_mlp(const uint4* wq, const float* esc, size_t e,
                   int r, int hi, int hi8, int lane,
                   __bf16* bH, __bf16* bH2,
                   int TB1, int TB2, int TB3, unsigned (*wpk)[4]) {
  v16bf aes0 = buildA8n(esc + e * 64 + hi * 8);
  v16bf aes1 = buildA8n(esc + e * 64 + 32 + hi * 8);
#pragma unroll
  for (int nt = 0; nt < 4; ++nt) {
    v16bf b0 = loadB(wq, TB1, 4, 0, nt, lane);
    v16bf b1 = loadB(wq, TB1, 4, 1, nt, lane);
    v8f c = wmmab(aes0, b0, v8zero());
    c = wmmab(aes1, b1, c);
#pragma unroll
    for (int v = 0; v < 8; ++v) c[v] = silu_beta(c[v] * INV8);
    storeC(bH, 72, nt * 16 + r, hi8, c, 1.0f);
  }
  wfence();
  {
    v16bf a0 = ldsA(bH, 72, r, 0, hi);
    v16bf a1 = ldsA(bH, 72, r, 32, hi);
#pragma unroll
    for (int nt = 0; nt < 4; ++nt) {
      v16bf b0 = loadB(wq, TB2, 4, 0, nt, lane);
      v16bf b1 = loadB(wq, TB2, 4, 1, nt, lane);
      v8f c = wmmab(a0, b0, v8zero());
      c = wmmab(a1, b1, c);
#pragma unroll
      for (int v = 0; v < 8; ++v) c[v] = silu_beta(c[v] * INV8);
      storeC(bH2, 72, nt * 16 + r, hi8, c, 1.0f);
    }
  }
  wfence();
  {
    v16bf a0 = ldsA(bH2, 72, r, 0, hi);
    v16bf a1 = ldsA(bH2, 72, r, 32, hi);
#pragma unroll
    for (int nt = 0; nt < 12; ++nt) {
      v16bf b0 = loadB(wq, TB3, 12, 0, nt, lane);
      v16bf b1 = loadB(wq, TB3, 12, 1, nt, lane);
      v8f c = wmmab(a0, b0, v8zero());
      c = wmmab(a1, b1, c);
#pragma unroll
      for (int k = 0; k < 4; ++k)
        wpk[nt][k] = pk_cvt(c[2*k] * INV8, c[2*k+1] * INV8);
    }
  }
}

// ---------------------------------------------------------------------------
// Weight prepack: f32 row-major [K][N] -> bf16 B-operand tile layout.
// ---------------------------------------------------------------------------
__global__ void prepack_kernel(const float* __restrict__ src,
                               unsigned short* __restrict__ dst, int K, int N) {
  int idx = blockIdx.x * blockDim.x + threadIdx.x;
  int total = K * N;
  if (idx >= total) return;
  int k = idx / N, n = idx - k * N;
  int kt = k >> 5, kl = k & 31;
  int nt = n >> 4, nl = n & 15;
  int hiq = (kl >> 3) & 1;
  int lane = nl + (hiq << 4);
  int j = (kl & 7) | ((kl >> 4) << 3);
  int ntiles = N >> 4;
  int off = ((kt * ntiles + nt) << 9) + (lane << 4) + j;
  dst[off] = f2bfbits(src[idx]);
}

// ---------------------------------------------------------------------------
// Main fused kernel. 256 threads = 8 waves; each wave owns 16 edges.
// ---------------------------------------------------------------------------
__global__ __launch_bounds__(256) void mpblock_fused(
    const float* __restrict__ nsrc, const float* __restrict__ ndst,
    const float* __restrict__ efeat, const float* __restrict__ lee,
    const float* __restrict__ esc, const uint4* __restrict__ wq,
    float* __restrict__ out) {
  // Per-wave bf16 staging (rows padded +8 elements for bank-conflict-free b128).
  __shared__ __align__(16) __bf16 sbuf[8 * 9088];
  __shared__ __align__(16) float  sy[8][16][4];

  const int tid = threadIdx.x;
  const int wvid = tid >> 5, lane = tid & 31;
  const int r = lane & 15, hi = lane >> 4, hi8 = hi << 3;
  const int e0 = blockIdx.x * 128 + wvid * 16;
  const size_t e = (size_t)(e0 + r);

  __bf16* Wb   = sbuf + wvid * 9088;
  __bf16* bH   = Wb + 0;      // 16 x 72  (mlp hidden 1)
  __bf16* bMSc = Wb + 1152;   // 16 x 40  (m_s 32-col chunk)
  __bf16* bMV  = Wb + 1792;   // 16 x 72  (gated m_v / mlp hidden 2)
  __bf16* bNS  = Wb + 2944;   // 16 x 72  (n_s)
  __bf16* bNV  = Wb + 4096;   // 3 x 16 x 40 (n_v)
  __bf16* bES  = Wb + 6016;   // 16 x 72  (e_s)
  __bf16* bEV  = Wb + 7168;   // 3 x 16 x 40 (e_v)

  // Stage ys/yv (both lane halves write identical data).
  *(float4*)&sy[wvid][r][0] = *(const float4*)(lee + e * 4);
  wfence();
  const float ysA  = sy[wvid][r][0];
  const float yvA0 = sy[wvid][r][1], yvA1 = sy[wvid][r][2], yvA2 = sy[wvid][r][3];
  float yC0[8], yC1[8], yC2[8];
#pragma unroll
  for (int v = 0; v < 8; ++v) {
    yC0[v] = sy[wvid][hi8 + v][1];
    yC1[v] = sy[wvid][hi8 + v][2];
    yC2[v] = sy[wvid][hi8 + v][3];
  }

  unsigned wpk[12][4];

  // ========================= NODE BRANCH =========================
  run_mlp(wq, esc, e, r, hi, hi8, lane, bH, bMV, TB_Wnm1, TB_Wnm2, TB_Wnm3, wpk);

  // t = xs @ Wn_sv / sqrt(128)   (16 x 32 C tiles)
  v8f tC[2];
  {
    v16bf Axs[4];
#pragma unroll
    for (int kt = 0; kt < 4; ++kt) {
      int k0 = kt * 32 + hi * 8;
      const float* p = (k0 < 64) ? nsrc + e * 160 + k0 : ndst + e * 160 + (k0 - 64);
      Axs[kt] = buildA8n(p);
    }
#pragma unroll
    for (int nt = 0; nt < 2; ++nt) {
      v8f c = v8zero();
#pragma unroll
      for (int kt = 0; kt < 4; ++kt) c = wmmab(Axs[kt], loadB(wq, TB_Wn_sv, 2, kt, nt, lane), c);
#pragma unroll
      for (int v = 0; v < 8; ++v) c[v] *= INV_SQRT128;
      tC[nt] = c;
    }
  }

  // Vector A panels: dot (1/sqrt(3)/8 folded) and xv_i * ys / 8.
  v16bf Adot[2], Axy[3][2];
#pragma unroll
  for (int kt = 0; kt < 2; ++kt) {
    const float* pv = ((kt == 0) ? nsrc : ndst) + e * 160 + 64;
    buildV(pv, hi * 8,      yvA0, yvA1, yvA2, INV_SQRT3 * INV8, ysA * INV8, 0,
           Adot[kt], Axy[0][kt], Axy[1][kt], Axy[2][kt]);
    buildV(pv, hi * 8 + 16, yvA0, yvA1, yvA2, INV_SQRT3 * INV8, ysA * INV8, 1,
           Adot[kt], Axy[0][kt], Axy[1][kt], Axy[2][kt]);
  }

  // Per-component: v_b GEMM, gate (v_a = t*yv, v_b) by w[:,128+u], n_v GEMM.
#pragma unroll
  for (int i = 0; i < 3; ++i) {
    v8f mvb[2];
#pragma unroll
    for (int nt = 0; nt < 2; ++nt) {
      v8f c = v8zero();
#pragma unroll
      for (int kt = 0; kt < 2; ++kt) c = wmmab(Axy[i][kt], loadB(wq, TB_Wn_vs, 2, kt, nt, lane), c);
      mvb[nt] = c;
    }
#pragma unroll
    for (int nt = 0; nt < 2; ++nt) {
#pragma unroll
      for (int k = 0; k < 4; ++k) {
        float ya = (i == 0) ? yC0[2*k]   : (i == 1) ? yC1[2*k]   : yC2[2*k];
        float yb = (i == 0) ? yC0[2*k+1] : (i == 1) ? yC1[2*k+1] : yC2[2*k+1];
        unsigned ma = pk_mul(pk_cvt(tC[nt][2*k] * ya, tC[nt][2*k+1] * yb), wpk[8 + nt][k]);
        unsigned mb = pk_mul(pk_cvt(mvb[nt][2*k], mvb[nt][2*k+1]), wpk[10 + nt][k]);
        storePair(bMV, 72, nt * 16 + r,      hi8 + 2*k, ma);
        storePair(bMV, 72, 32 + nt * 16 + r, hi8 + 2*k, mb);
      }
    }
    wfence();
    {
      v16bf a0 = ldsA(bMV, 72, r, 0, hi);
      v16bf a1 = ldsA(bMV, 72, r, 32, hi);
#pragma unroll
      for (int nt = 0; nt < 2; ++nt) {
        v8f c = wmmab(a0, loadB(wq, TB_Wns_v, 2, 0, nt, lane), v8zero());
        c = wmmab(a1, loadB(wq, TB_Wns_v, 2, 1, nt, lane), c);
        storeC(bNV + i * 640, 40, nt * 16 + r, hi8, c, INV8);
      }
    }
    wfence();
  }

  // Scalar path: m_s tiles (s_a | s_b) -> gate -> accumulate n_s per K-chunk.
  {
    v16bf Axsy[4];
#pragma unroll
    for (int kt = 0; kt < 4; ++kt) {
      int k0 = kt * 32 + hi * 8;
      const float* p = (k0 < 64) ? nsrc + e * 160 + k0 : ndst + e * 160 + (k0 - 64);
      Axsy[kt] = buildA8(p, ysA * INV_SQRT128);
    }
    v8f nsC[4] = {v8zero(), v8zero(), v8zero(), v8zero()};
#pragma unroll
    for (int kc = 0; kc < 4; ++kc) {
#pragma unroll
      for (int t2 = 0; t2 < 2; ++t2) {
        const int nt = kc * 2 + t2;
        v8f c = v8zero();
        if (nt < 4) {
#pragma unroll
          for (int kt = 0; kt < 4; ++kt) c = wmmab(Axsy[kt], loadB(wq, TB_Wn_ss, 4, kt, nt, lane), c);
        } else {
#pragma unroll
          for (int kt = 0; kt < 2; ++kt) c = wmmab(Adot[kt], loadB(wq, TB_Wn_vv, 4, kt, nt - 4, lane), c);
        }
        gateStore(bMSc, 40, t2 * 16 + r, hi8, c, wpk[nt]);
      }
      wfence();
      {
        v16bf am = ldsA(bMSc, 40, r, 0, hi);
#pragma unroll
        for (int nt = 0; nt < 4; ++nt)
          nsC[nt] = wmmab(am, loadB(wq, TB_Wns_s, 4, kc, nt, lane), nsC[nt]);
      }
      wfence();
    }
#pragma unroll
    for (int nt = 0; nt < 4; ++nt) storeC(bNS, 72, nt * 16 + r, hi8, nsC[nt], INV_SQRT128);
    wfence();
  }

  // ========================= EDGE BRANCH =========================
  run_mlp(wq, esc, e, r, hi, hi8, lane, bH, bMV, TB_Wem1, TB_Wem2, TB_Wem3, wpk);

  // t_e = fs @ We_sv / 8
  v8f teC[2];
  {
    v16bf Afs0 = buildA8n(efeat + e * 160 + hi * 8);
    v16bf Afs1 = buildA8n(efeat + e * 160 + 32 + hi * 8);
#pragma unroll
    for (int nt = 0; nt < 2; ++nt) {
      v8f c = wmmab(Afs0, loadB(wq, TB_We_sv, 2, 0, nt, lane), v8zero());
      c = wmmab(Afs1, loadB(wq, TB_We_sv, 2, 1, nt, lane), c);
#pragma unroll
      for (int v = 0; v < 8; ++v) c[v] *= INV8;
      teC[nt] = c;
    }
  }

  v16bf Fdot, Fxy0, Fxy1, Fxy2;
  {
    const float* pv = efeat + e * 160 + 64;
    buildV(pv, hi * 8,      yvA0, yvA1, yvA2, INV_SQRT3 * INV_SQRT32, ysA * INV_SQRT32, 0, Fdot, Fxy0, Fxy1, Fxy2);
    buildV(pv, hi * 8 + 16, yvA0, yvA1, yvA2, INV_SQRT3 * INV_SQRT32, ysA * INV_SQRT32, 1, Fdot, Fxy0, Fxy1, Fxy2);
  }

#pragma unroll
  for (int i = 0; i < 3; ++i) {
    v16bf Fi = (i == 0) ? Fxy0 : (i == 1) ? Fxy1 : Fxy2;
    v8f mvb[2];
#pragma unroll
    for (int nt = 0; nt < 2; ++nt) mvb[nt] = wmmab(Fi, loadB(wq, TB_We_vs, 2, 0, nt, lane), v8zero());
#pragma unroll
    for (int nt = 0; nt < 2; ++nt) {
#pragma unroll
      for (int k = 0; k < 4; ++k) {
        float ya = (i == 0) ? yC0[2*k]   : (i == 1) ? yC1[2*k]   : yC2[2*k];
        float yb = (i == 0) ? yC0[2*k+1] : (i == 1) ? yC1[2*k+1] : yC2[2*k+1];
        unsigned ma = pk_mul(pk_cvt(teC[nt][2*k] * ya, teC[nt][2*k+1] * yb), wpk[8 + nt][k]);
        unsigned mb = pk_mul(pk_cvt(mvb[nt][2*k], mvb[nt][2*k+1]), wpk[10 + nt][k]);
        storePair(bMV, 72, nt * 16 + r,      hi8 + 2*k, ma);
        storePair(bMV, 72, 32 + nt * 16 + r, hi8 + 2*k, mb);
      }
    }
    wfence();
    {
      v16bf a0 = ldsA(bMV, 72, r, 0, hi);
      v16bf a1 = ldsA(bMV, 72, r, 32, hi);
#pragma unroll
      for (int nt = 0; nt < 2; ++nt) {
        v8f c = wmmab(a0, loadB(wq, TB_Wes_v, 2, 0, nt, lane), v8zero());
        c = wmmab(a1, loadB(wq, TB_Wes_v, 2, 1, nt, lane), c);
        storeC(bEV + i * 640, 40, nt * 16 + r, hi8, c, INV8);
      }
    }
    wfence();
  }

  {
    v16bf Afsy0 = buildA8(efeat + e * 160 + hi * 8, ysA * INV8);
    v16bf Afsy1 = buildA8(efeat + e * 160 + 32 + hi * 8, ysA * INV8);
    v8f esC[4] = {v8zero(), v8zero(), v8zero(), v8zero()};
#pragma unroll
    for (int kc = 0; kc < 4; ++kc) {
#pragma unroll
      for (int t2 = 0; t2 < 2; ++t2) {
        const int nt = kc * 2 + t2;
        v8f c;
        if (nt < 4) {
          c = wmmab(Afsy0, loadB(wq, TB_We_ss, 4, 0, nt, lane), v8zero());
          c = wmmab(Afsy1, loadB(wq, TB_We_ss, 4, 1, nt, lane), c);
        } else {
          c = wmmab(Fdot, loadB(wq, TB_We_vv, 4, 0, nt - 4, lane), v8zero());
        }
        gateStore(bMSc, 40, t2 * 16 + r, hi8, c, wpk[nt]);
      }
      wfence();
      {
        v16bf am = ldsA(bMSc, 40, r, 0, hi);
#pragma unroll
        for (int nt = 0; nt < 4; ++nt)
          esC[nt] = wmmab(am, loadB(wq, TB_Wes_s, 4, kc, nt, lane), esC[nt]);
      }
      wfence();
    }
#pragma unroll
    for (int nt = 0; nt < 4; ++nt) storeC(bES, 72, nt * 16 + r, hi8, esC[nt], INV_SQRT128);
    wfence();
  }

  // ========================= FINAL LINEAR + STORE =========================
  {
    v16bf an0 = ldsA(bNS, 72, r, 0, hi);
    v16bf an1 = ldsA(bNS, 72, r, 32, hi);
    v16bf ae0 = ldsA(bES, 72, r, 0, hi);
    v16bf ae1 = ldsA(bES, 72, r, 32, hi);
#pragma unroll
    for (int nt = 0; nt < 4; ++nt) {
      v8f c = wmmab(an0, loadB(wq, TB_Wno_s, 4, 0, nt, lane), v8zero());
      c = wmmab(an1, loadB(wq, TB_Wno_s, 4, 1, nt, lane), c);
      c = wmmab(ae0, loadB(wq, TB_Weo_s, 4, 0, nt, lane), c);
      c = wmmab(ae1, loadB(wq, TB_Weo_s, 4, 1, nt, lane), c);
#pragma unroll
      for (int v = 0; v < 8; ++v)
        out[(size_t)(e0 + hi8 + v) * 160 + nt * 16 + r] = c[v] * INV8;
    }
  }
#pragma unroll
  for (int i = 0; i < 3; ++i) {
    v16bf an = ldsA(bNV + i * 640, 40, r, 0, hi);
    v16bf ae = ldsA(bEV + i * 640, 40, r, 0, hi);
#pragma unroll
    for (int nt = 0; nt < 2; ++nt) {
      v8f c = wmmab(an, loadB(wq, TB_Wno_v, 2, 0, nt, lane), v8zero());
      c = wmmab(ae, loadB(wq, TB_Weo_v, 2, 0, nt, lane), c);
#pragma unroll
      for (int v = 0; v < 8; ++v)
        out[(size_t)(e0 + hi8 + v) * 160 + 64 + 3 * (nt * 16 + r) + i] = c[v] * INV_SQRT32;
    }
  }
}

// ---------------------------------------------------------------------------
// Host launcher
// ---------------------------------------------------------------------------
struct WInfo { int idx; int K; int N; long off; };
static const WInfo g_winfo[22] = {
  {5,128,64,0},      {6,64,64,8192},   {7,128,32,12288}, {8,64,32,16384},
  {9,64,64,18432},   {10,32,64,22528}, {11,64,32,24576}, {12,32,32,26624},
  {13,128,64,27648}, {14,64,32,35840}, {15,128,64,37888},{16,64,32,46080},
  {17,64,64,48128},  {18,32,32,52224}, {19,64,64,53248}, {20,32,32,57344},
  {21,64,64,58368},  {22,64,64,62464}, {23,64,192,66560},
  {24,64,64,78848},  {25,64,64,82944}, {26,64,192,87040}
};

extern "C" void kernel_launch(void* const* d_in, const int* in_sizes, int n_in,
                              void* d_out, int out_size, void* d_ws, size_t ws_size,
                              hipStream_t stream) {
  (void)in_sizes; (void)n_in; (void)out_size;
  if (ws_size < (size_t)99328 * 2) return;  // need 194 KB of scratch for packed weights

  unsigned short* wpack = (unsigned short*)d_ws;
  for (int i = 0; i < 22; ++i) {
    const WInfo& w = g_winfo[i];
    int total = w.K * w.N;
    prepack_kernel<<<dim3((total + 255) / 256), 256, 0, stream>>>(
        (const float*)d_in[w.idx], wpack + w.off, w.K, w.N);
  }

  mpblock_fused<<<dim3(1024), 256, 0, stream>>>(
      (const float*)d_in[0], (const float*)d_in[1], (const float*)d_in[2],
      (const float*)d_in[3], (const float*)d_in[4],
      (const uint4*)d_ws, (float*)d_out);
}
